// SelfAttentionNarrow_15504831939288
// MI455X (gfx1250) — matrix operand order repaired and checked by reading, not compiled, not run on hardware
//
#include <hip/hip_runtime.h>

typedef __attribute__((ext_vector_type(16))) _Float16 v16h;
typedef __attribute__((ext_vector_type(8)))  float    v8f;

constexpr int B_ = 8, T_ = 1024, E_ = 1024, H_ = 16, S_ = 64;

__device__ __forceinline__ v8f wmma_f16(const v16h a, const v16h b, v8f c) {
  // D = A(16x32 f16) * B(32x16 f16) + C(16x16 f32)
  return __builtin_amdgcn_wmma_f32_16x16x32_f16(false, a, false, b, (short)0, c,
                                                false, false);
}

// LDS byte offset (wave-relative) of a __shared__ object, for async-to-LDS VDST.
__device__ __forceinline__ unsigned lds_off(const void* p) {
  return (unsigned)(unsigned long long)(__attribute__((address_space(3))) const void*)p;
}

// Butterfly reductions confined to 16-lane halves (xor of bits 0..3 never
// crosses bit 4), matching the C/D-layout row ownership (rows 0-7 in lanes
// 0-15, rows 8-15 in lanes 16-31).
__device__ __forceinline__ float half_max(float v) {
  v = fmaxf(v, __shfl_xor(v, 1, 32));
  v = fmaxf(v, __shfl_xor(v, 2, 32));
  v = fmaxf(v, __shfl_xor(v, 4, 32));
  v = fmaxf(v, __shfl_xor(v, 8, 32));
  return v;
}
__device__ __forceinline__ float half_sum(float v) {
  v += __shfl_xor(v, 1, 32);
  v += __shfl_xor(v, 2, 32);
  v += __shfl_xor(v, 4, 32);
  v += __shfl_xor(v, 8, 32);
  return v;
}

// ---------------------------------------------------------------------------
// Kernel 1: per-head Q/K/V projections.  Each wave does a 16-row x-tile
// against the shared 64x64 weights (f16, staged in LDS).  Q gets the full
// e^-0.5 = 1/32 score scale folded in.  V is written channel-major
// (transposed) so the attention kernel gets contiguous B-fragments.
// ---------------------------------------------------------------------------
__global__ __launch_bounds__(128) void proj_kernel(
    const float* __restrict__ x, const float* __restrict__ Wk,
    const float* __restrict__ Wq, const float* __restrict__ Wv,
    _Float16* __restrict__ Qh, _Float16* __restrict__ Kh,
    _Float16* __restrict__ Vth) {
  __shared__ _Float16 sWq[S_ * S_];
  __shared__ _Float16 sWk[S_ * S_];
  __shared__ _Float16 sWv[S_ * S_];
  const int tid  = threadIdx.x;
  const int bh   = blockIdx.x >> 4;   // (b*H + h)
  const int tblk = blockIdx.x & 15;   // 16 tiles of 64 rows along T
  const int b = bh >> 4, h = bh & 15;

  for (int i = tid; i < S_ * S_; i += 128) {
    sWq[i] = (_Float16)(Wq[i] * 0.03125f);  // fold E^-0.5 into Q
    sWk[i] = (_Float16)Wk[i];
    sWv[i] = (_Float16)Wv[i];
  }
  __syncthreads();

  const int wave  = tid >> 5, lane = tid & 31;
  const int hi    = (lane >> 4) & 1;
  const int nlane = lane & 15;
  const int kb    = hi ? 8 : 0;    // A-frag K base
  const int sb    = hi ? 16 : 0;   // unused; B-frags use kb too
  const int t0    = tblk * 64 + wave * 16;

  // A fragments: x row (t0+nlane), channels h*64..h*64+63, f32 -> f16.
  const float* xr = x + ((size_t)(b * T_ + t0 + nlane)) * E_ + h * S_;
  v16h a0, a1;
#pragma unroll
  for (int i = 0; i < 8; ++i) {
    a0[i]     = (_Float16)xr[kb + i];
    a0[i + 8] = (_Float16)xr[kb + 16 + i];
    a1[i]     = (_Float16)xr[32 + kb + i];
    a1[i + 8] = (_Float16)xr[32 + kb + 16 + i];
  }

  auto do_proj = [&](const _Float16* Ws, _Float16* dst, bool transposed) {
#pragma unroll
    for (int nt = 0; nt < 4; ++nt) {
      const _Float16* wr = Ws + (size_t)(nt * 16 + nlane) * S_;  // row o of W
      v16h b0, b1;
#pragma unroll
      for (int i = 0; i < 8; ++i) {
        b0[i]     = wr[kb + i];        // B[k=s][n=o] = W[o][s], s = 0..31
        b0[i + 8] = wr[kb + 16 + i];
        b1[i]     = wr[32 + kb + i];   // s = 32..63
        b1[i + 8] = wr[32 + kb + 16 + i];
      }
      v8f c = {};
      c = wmma_f16(a0, b0, c);
      c = wmma_f16(a1, b1, c);
      const int col = nt * 16 + nlane;
#pragma unroll
      for (int r = 0; r < 8; ++r) {
        const int trow = t0 + r + hi * 8;
        if (!transposed)
          dst[((size_t)bh * T_ + trow) * S_ + col] = (_Float16)c[r];
        else
          dst[((size_t)bh * S_ + col) * T_ + trow] = (_Float16)c[r];
      }
    }
  };
  do_proj(sWq, Qh, false);
  do_proj(sWk, Kh, false);
  do_proj(sWv, Vth, true);
}

// ---------------------------------------------------------------------------
// Kernel 2: flash attention.  4 waves/WG, each wave owns a 16-query tile.
// K (row-major) and V^T (channel-major) 32-key blocks are streamed into
// double-buffered LDS tiles with GLOBAL_LOAD_ASYNC_TO_LDS_B128 (ASYNCcnt),
// prefetching block i+1 while block i feeds the WMMA chain.  Online softmax
// with running (max, sum); P round-trips through LDS for C->A layout.
// ---------------------------------------------------------------------------
__global__ __launch_bounds__(128) void attn_kernel(
    const _Float16* __restrict__ Qh, const _Float16* __restrict__ Kh,
    const _Float16* __restrict__ Vth, const int* __restrict__ lengths,
    _Float16* __restrict__ Oh) {
  __shared__ _Float16 sK[2][32 * 64];    // [buf][key][chan]
  __shared__ _Float16 sV[2][64 * 32];    // [buf][chan][key]
  __shared__ _Float16 sP[4 * 16 * 32];   // per-wave P tile

  const int tid  = threadIdx.x;
  const int bh   = blockIdx.x >> 4;
  const int qblk = blockIdx.x & 15;
  const int b = bh >> 4, h = bh & 15;
  const int wave = tid >> 5, lane = tid & 31;
  const int hi = (lane >> 4) & 1, nlane = lane & 15;
  const int kb = hi ? 8 : 0, sb = hi ? 16 : 0;
  const int q0wg = qblk * 64;
  const int q0   = q0wg + wave * 16;
  const int len  = lengths[b];

  // Q A-fragments for this wave's 16 query rows (scale already folded in).
  const _Float16* qr = Qh + ((size_t)bh * T_ + q0 + nlane) * S_;
  v16h qa0, qa1;
#pragma unroll
  for (int i = 0; i < 8; ++i) {
    qa0[i]     = qr[kb + i];
    qa0[i + 8] = qr[kb + 16 + i];
    qa1[i]     = qr[32 + kb + i];
    qa1[i + 8] = qr[32 + kb + 16 + i];
  }

  v8f oacc[4] = {{}, {}, {}, {}};
  float mrun[8], lrun[8];
#pragma unroll
  for (int r = 0; r < 8; ++r) { mrun[r] = -1e30f; lrun[r] = 0.f; }

  _Float16* Pw = sP + wave * (16 * 32);
  const int nblk = qblk * 2 + 2;  // causal bound: keys <= q0wg+63

  // Async stage of one 32-key block into LDS buffer `buf`.
  // Each thread copies a 32B stripe of K and of V^T (2x b128 each); the
  // instruction's IOFFSET applies to both the LDS and global address.
  auto stage = [&](int blk, int buf) {
    const int k0   = blk * 32;
    const int base = tid * 16;
    const int krow = base >> 6, kcol = base & 63;
    const _Float16* ks = Kh + ((size_t)bh * T_ + k0 + krow) * S_ + kcol;
    const int vrow = base >> 5, vcol = base & 31;
    const _Float16* vs = Vth + ((size_t)bh * S_ + vrow) * T_ + k0 + vcol;
    const unsigned dK = lds_off(&sK[buf][base]);
    const unsigned dV = lds_off(&sV[buf][base]);
    asm volatile(
        "global_load_async_to_lds_b128 %0, %2, off\n\t"
        "global_load_async_to_lds_b128 %0, %2, off offset:16\n\t"
        "global_load_async_to_lds_b128 %1, %3, off\n\t"
        "global_load_async_to_lds_b128 %1, %3, off offset:16"
        :: "v"(dK), "v"(dV),
           "v"((unsigned long long)ks), "v"((unsigned long long)vs)
        : "memory");
  };

  stage(0, 0);
  asm volatile("s_wait_asynccnt 0x0" ::: "memory");
  __syncthreads();

  for (int blk = 0; blk < nblk; ++blk) {
    const int buf = blk & 1;
    const int k0  = blk * 32;
    if (blk + 1 < nblk) stage(blk + 1, buf ^ 1);  // prefetch next block

    if (k0 <= q0 + 15) {  // wave-uniform: some key in block is unmasked
      // scores: S(16x32) = Q(16x64) @ K_blk^T, two 16-key subtiles
      v8f sc[2];
#pragma unroll
      for (int st = 0; st < 2; ++st) {
        const _Float16* kr = &sK[buf][(st * 16 + nlane) * 64];
        v16h b0, b1;
#pragma unroll
        for (int i = 0; i < 8; ++i) {
          b0[i]     = kr[kb + i];
          b0[i + 8] = kr[kb + 16 + i];
          b1[i]     = kr[32 + kb + i];
          b1[i + 8] = kr[32 + kb + 16 + i];
        }
        v8f c = {};
        c = wmma_f16(qa0, b0, c);
        c = wmma_f16(qa1, b1, c);
#pragma unroll
        for (int r = 0; r < 8; ++r) {  // causal + length mask
          const int qrow = q0 + r + hi * 8;
          const int kcol = k0 + st * 16 + nlane;
          if (kcol > qrow || kcol >= len) c[r] = -1e30f;
        }
        sc[st] = c;
      }

      float alpha[8];
#pragma unroll
      for (int r = 0; r < 8; ++r) {
        const float rm   = half_max(fmaxf(sc[0][r], sc[1][r]));
        const float mnew = fmaxf(mrun[r], rm);
        alpha[r] = __expf(mrun[r] - mnew);
        mrun[r]  = mnew;
      }
#pragma unroll
      for (int r = 0; r < 8; ++r) {
        const float p0 = __expf(sc[0][r] - mrun[r]);
        const float p1 = __expf(sc[1][r] - mrun[r]);
        lrun[r] = lrun[r] * alpha[r] + half_sum(p0 + p1);
        const int prow = r + hi * 8;
        Pw[prow * 32 + nlane]      = (_Float16)p0;
        Pw[prow * 32 + 16 + nlane] = (_Float16)p1;
      }
#pragma unroll
      for (int nt = 0; nt < 4; ++nt)
#pragma unroll
        for (int r = 0; r < 8; ++r) oacc[nt][r] *= alpha[r];

      asm volatile("s_wait_dscnt 0x0" ::: "memory");  // P stores -> A-frag loads

      // P A-fragment (16x32, K=32)
      const _Float16* pr = Pw + nlane * 32;
      v16h pa;
#pragma unroll
      for (int i = 0; i < 8; ++i) {
        pa[i]     = pr[kb + i];
        pa[i + 8] = pr[kb + 16 + i];
      }
      // O += P @ V_blk  (4 channel subtiles)
#pragma unroll
      for (int nt = 0; nt < 4; ++nt) {
        const _Float16* vr = &sV[buf][(nt * 16 + nlane) * 32];
        v16h vb;
#pragma unroll
        for (int i = 0; i < 8; ++i) { vb[i] = vr[kb + i]; vb[i + 8] = vr[kb + 16 + i]; }
        oacc[nt] = wmma_f16(pa, vb, oacc[nt]);
      }
    }

    // Make the prefetched buffer visible to all waves before next iteration.
    asm volatile("s_wait_asynccnt 0x0" ::: "memory");
    __syncthreads();
  }

  // epilogue: divide by softmax denominator, store f16 merged (b,t,e) layout
#pragma unroll
  for (int nt = 0; nt < 4; ++nt) {
    const int col = h * S_ + nt * 16 + nlane;
#pragma unroll
    for (int r = 0; r < 8; ++r) {
      const int trow = q0 + r + hi * 8;
      Oh[((size_t)b * T_ + trow) * E_ + col] = (_Float16)(oacc[nt][r] / lrun[r]);
    }
  }
}

// ---------------------------------------------------------------------------
// Kernel 3: output projection  y = O(8192x1024) @ Wu^T(1024x1024) + bu.
// WG = 64x64 output tile, 4 waves along M; Wu staged/converted to f16 in LDS
// per 32-wide K chunk and shared by all 4 waves.
// ---------------------------------------------------------------------------
__global__ __launch_bounds__(128) void outproj_kernel(
    const _Float16* __restrict__ Oh, const float* __restrict__ Wu,
    const float* __restrict__ bu, float* __restrict__ out) {
  __shared__ _Float16 sW[64 * 32];  // [o_local][k_local]
  const int tid = threadIdx.x;
  const int mt  = blockIdx.x >> 4;   // 128 M tiles
  const int ntw = blockIdx.x & 15;   // 16 N tiles
  const int wave = tid >> 5, lane = tid & 31;
  const int hi = (lane >> 4) & 1, nlane = lane & 15;
  const int kb = hi ? 8 : 0, sb = hi ? 16 : 0;
  const int m0 = mt * 64 + wave * 16;
  const int n0 = ntw * 64;

  v8f acc[4] = {{}, {}, {}, {}};
  for (int k0 = 0; k0 < E_; k0 += 32) {
    {  // stage Wu[n0..n0+63][k0..k0+31] as f16
      const int base = tid * 16;
      const int ol = base >> 5, kk = base & 31;
      const float* src = Wu + ((size_t)(n0 + ol)) * E_ + k0 + kk;
#pragma unroll
      for (int i = 0; i < 16; ++i) sW[base + i] = (_Float16)src[i];
    }
    __syncthreads();

    const _Float16* orow = Oh + (size_t)(m0 + nlane) * E_ + k0;
    v16h a;
#pragma unroll
    for (int i = 0; i < 8; ++i) {
      a[i]     = orow[kb + i];
      a[i + 8] = orow[kb + 16 + i];
    }
#pragma unroll
    for (int nt = 0; nt < 4; ++nt) {
      const _Float16* wr = sW + (size_t)(nt * 16 + nlane) * 32;
      v16h bf;
#pragma unroll
      for (int i = 0; i < 8; ++i) { bf[i] = wr[kb + i]; bf[i + 8] = wr[kb + 16 + i]; }
      acc[nt] = wmma_f16(a, bf, acc[nt]);
    }
    __syncthreads();
  }

#pragma unroll
  for (int nt = 0; nt < 4; ++nt) {
    const int col  = n0 + nt * 16 + nlane;
    const float bb = bu[col];
#pragma unroll
    for (int r = 0; r < 8; ++r) {
      const int row = m0 + r + hi * 8;
      out[(size_t)row * E_ + col] = acc[nt][r] + bb;
    }
  }
}

extern "C" void kernel_launch(void* const* d_in, const int* in_sizes, int n_in,
                              void* d_out, int out_size, void* d_ws,
                              size_t ws_size, hipStream_t stream) {
  (void)in_sizes; (void)n_in; (void)out_size; (void)ws_size;
  // setup_inputs() order: x, lengths, Wk, Wq, Wv, Wu, bu
  const float* x       = (const float*)d_in[0];
  const int*   lengths = (const int*)d_in[1];
  const float* Wk      = (const float*)d_in[2];
  const float* Wq      = (const float*)d_in[3];
  const float* Wv      = (const float*)d_in[4];
  const float* Wu      = (const float*)d_in[5];
  const float* bu      = (const float*)d_in[6];
  float* out = (float*)d_out;

  const size_t nqkv = (size_t)B_ * H_ * T_ * S_;  // 8Mi halfs each
  _Float16* Qh  = (_Float16*)d_ws;   // 16 MB
  _Float16* Kh  = Qh + nqkv;         // 16 MB
  _Float16* Vth = Kh + nqkv;         // 16 MB (channel-major)
  _Float16* Oh  = Vth + nqkv;        // 16 MB (merged b,t,e)

  const dim3 blk(128);
  proj_kernel<<<dim3(B_ * H_ * (T_ / 64)), blk, 0, stream>>>(x, Wk, Wq, Wv, Qh,
                                                             Kh, Vth);
  attn_kernel<<<dim3(B_ * H_ * (T_ / 64)), blk, 0, stream>>>(Qh, Kh, Vth,
                                                             lengths, Oh);
  outproj_kernel<<<dim3((B_ * T_ / 64) * (E_ / 64)), blk, 0, stream>>>(Oh, Wu,
                                                                       bu, out);
}
